// GUNet_10694468567466
// MI455X (gfx1250) — matrix operand order repaired and checked by reading, not compiled, not run on hardware
//
#include <hip/hip_runtime.h>

// ---------------------------------------------------------------- types
typedef _Float16 v16h  __attribute__((ext_vector_type(16)));
typedef _Float16 half8 __attribute__((ext_vector_type(8)));
typedef _Float16 half4 __attribute__((ext_vector_type(4)));
typedef float    v8f   __attribute__((ext_vector_type(8)));

#define BM 128
#define BN 128
#define BK 32
#define BKP 40   // padded LDS row stride (halfs); keeps all frag chunks 16B aligned

// ---------------------------------------------------------------- WMMA GEMM
// C[M,N] = A[M,K] @ B[K,N], all row-major contiguous f32; f16 staged, f32 acc.
__global__ __launch_bounds__(256) void gemm_f16wmma(
    const float* __restrict__ A, const float* __restrict__ B,
    float* __restrict__ C, int M, int N, int K)
{
    __shared__ __align__(16) _Float16 sA[BM * BKP];
    __shared__ __align__(16) _Float16 sB[BN * BKP];   // stored N-major: sB[n][k]

    const int tid  = threadIdx.x;
    const int lane = tid & 31;
    const int wave = tid >> 5;      // 0..7
    const int wm   = wave >> 1;     // 0..3 : 32-row wave tiles
    const int wn   = wave & 1;      // 0..1 : 64-col wave tiles

    const int blockM = blockIdx.y * BM;
    const int blockN = blockIdx.x * BN;

    v8f acc[2][4] = {};

    // per-lane fragment coordinates (wave32, 16-bit A / B; contiguous k = 16*(lane/16) + s, same order for A and B)
    const int mA = lane & 15;
    const int kA = (lane < 16) ? 0 : 16;   // A: halfs [kA..kA+15]
    const int nB = lane & 15;
    const int kB = (lane < 16) ? 0 : 16;   // B: halfs [kB..kB+15]

    const bool fullMN   = (blockM + BM <= M) && (blockN + BN <= N);
    const bool aligned4 = ((K & 3) == 0) && ((N & 3) == 0);

    // staging coordinates (fast path), hoisted
    const int arow = tid >> 3;              // A row in tile (+32 per it)
    const int akq  = (tid & 7) << 2;        // A k group of 4
    const int bkk  = tid >> 5;              // B k row (+8 per it)
    const int bnq  = (tid & 31) << 2;       // B n group of 4

    for (int k0 = 0; k0 < K; k0 += BK) {
        if (fullMN && aligned4 && (k0 + BK <= K)) {
            // ------- fast staging: batch all global loads, then convert/store
            float4 va0, va1, va2, va3, vb0, vb1, vb2, vb3;
            {
                const float* pa = &A[(long)(blockM + arow) * K + k0 + akq];
                const long strideA = (long)32 * K;   // +32 rows per it
                va0 = *reinterpret_cast<const float4*>(pa);
                va1 = *reinterpret_cast<const float4*>(pa + strideA);
                va2 = *reinterpret_cast<const float4*>(pa + 2 * strideA);
                va3 = *reinterpret_cast<const float4*>(pa + 3 * strideA);
                const float* pb = &B[(long)(k0 + bkk) * N + blockN + bnq];
                const long strideB = (long)8 * N;    // +8 k-rows per it
                vb0 = *reinterpret_cast<const float4*>(pb);
                vb1 = *reinterpret_cast<const float4*>(pb + strideB);
                vb2 = *reinterpret_cast<const float4*>(pb + 2 * strideB);
                vb3 = *reinterpret_cast<const float4*>(pb + 3 * strideB);
            }
            {
                float4 va[4] = { va0, va1, va2, va3 };
                #pragma unroll
                for (int it = 0; it < 4; ++it) {
                    half4 h = { (_Float16)va[it].x, (_Float16)va[it].y,
                                (_Float16)va[it].z, (_Float16)va[it].w };
                    *reinterpret_cast<half4*>(&sA[(arow + it * 32) * BKP + akq]) = h;
                }
                float4 vb[4] = { vb0, vb1, vb2, vb3 };
                #pragma unroll
                for (int it = 0; it < 4; ++it) {
                    int kk = bkk + it * 8;
                    sB[(bnq + 0) * BKP + kk] = (_Float16)vb[it].x;
                    sB[(bnq + 1) * BKP + kk] = (_Float16)vb[it].y;
                    sB[(bnq + 2) * BKP + kk] = (_Float16)vb[it].z;
                    sB[(bnq + 3) * BKP + kk] = (_Float16)vb[it].w;
                }
            }
            // prefetch next K-tile (one per 128B cacheline) -> global_prefetch_b8
            if (k0 + 2 * BK <= K) {
                if (tid < 128) {
                    __builtin_prefetch(&A[(long)(blockM + tid) * K + k0 + BK], 0, 0);
                } else {
                    int t  = tid - 128;
                    int kk = t >> 2;
                    int nq = (t & 3) << 5;
                    __builtin_prefetch(&B[(long)(k0 + BK + kk) * N + blockN + nq], 0, 0);
                }
            }
        } else {
            // ---------------- guarded staging: branchless (clamped address) --
            #pragma unroll
            for (int idx = tid; idx < BM * BK; idx += 256) {
                int r  = idx >> 5;
                int kk = idx & 31;
                int gr = blockM + r, gk = k0 + kk;
                bool ok = (gr < M) && (gk < K);
                float v = A[ok ? (long)gr * K + gk : 0];
                sA[r * BKP + kk] = (_Float16)(ok ? v : 0.0f);
            }
            #pragma unroll
            for (int idx = tid; idx < BK * BN; idx += 256) {
                int kk = idx >> 7;
                int n  = idx & 127;
                int gk = k0 + kk, gn = blockN + n;
                bool ok = (gk < K) && (gn < N);
                float v = B[ok ? (long)gk * N + gn : 0];
                sB[n * BKP + kk] = (_Float16)(ok ? v : 0.0f);
            }
        }
        __syncthreads();

        v16h af[2], bf[4];
        #pragma unroll
        for (int mi = 0; mi < 2; ++mi) {
            const _Float16* pa = &sA[(wm * 32 + mi * 16 + mA) * BKP + kA];
            half8 lo = *reinterpret_cast<const half8*>(pa);
            half8 hi = *reinterpret_cast<const half8*>(pa + 8);
            af[mi] = __builtin_shufflevector(lo, hi,
                     0,1,2,3,4,5,6,7,8,9,10,11,12,13,14,15);
        }
        #pragma unroll
        for (int ni = 0; ni < 4; ++ni) {
            const _Float16* pb = &sB[(wn * 64 + ni * 16 + nB) * BKP + kB];
            half8 lo = *reinterpret_cast<const half8*>(pb);
            half8 hi = *reinterpret_cast<const half8*>(pb + 8);
            bf[ni] = __builtin_shufflevector(lo, hi,
                     0,1,2,3,4,5,6,7,8,9,10,11,12,13,14,15);
        }
        #pragma unroll
        for (int mi = 0; mi < 2; ++mi)
            #pragma unroll
            for (int ni = 0; ni < 4; ++ni)
                acc[mi][ni] = __builtin_amdgcn_wmma_f32_16x16x32_f16(
                    false, af[mi], false, bf[ni], (short)0, acc[mi][ni],
                    false, false);
        __syncthreads();
    }

    // store C  (C/D layout: VGPR g = row g (lanes 0-15) / row g+8 (lanes 16-31))
    const int cN   = lane & 15;
    const int rOff = (lane < 16) ? 0 : 8;
    if (fullMN) {
        #pragma unroll
        for (int mi = 0; mi < 2; ++mi)
            #pragma unroll
            for (int ni = 0; ni < 4; ++ni)
                #pragma unroll
                for (int g = 0; g < 8; ++g) {
                    int row = blockM + wm * 32 + mi * 16 + rOff + g;
                    int col = blockN + wn * 64 + ni * 16 + cN;
                    C[(long)row * N + col] = acc[mi][ni][g];
                }
    } else {
        #pragma unroll
        for (int mi = 0; mi < 2; ++mi)
            #pragma unroll
            for (int ni = 0; ni < 4; ++ni)
                #pragma unroll
                for (int g = 0; g < 8; ++g) {
                    int row = blockM + wm * 32 + mi * 16 + rOff + g;
                    int col = blockN + wn * 64 + ni * 16 + cN;
                    if (row < M && col < N) C[(long)row * N + col] = acc[mi][ni][g];
                }
    }
}

// ---------------------------------------------------------------- elementwise
__global__ void fill_zero_k(float* p, long n) {
    long i = (long)blockIdx.x * blockDim.x + threadIdx.x;
    if (i < n) p[i] = 0.0f;
}

__global__ void scatter_edges_k(const int* __restrict__ ei, float* A, int E, int n) {
    int e = blockIdx.x * blockDim.x + threadIdx.x;
    if (e < E) {
        int s = ei[e], d = ei[E + e];
        A[(long)s * n + d] = 1.0f;
        A[(long)d * n + s] = 1.0f;
    }
}

// dis[i] = rsqrt(rowsum(A)+1)   (deg of A+I; always > 0)
__global__ void row_dis_k(const float* __restrict__ A, int n, float* dis) {
    __shared__ float red[256];
    int i = blockIdx.x;
    float s = 0.0f;
    for (int j = threadIdx.x; j < n; j += 256) s += A[(long)i * n + j];
    red[threadIdx.x] = s;
    __syncthreads();
    for (int k = 128; k > 0; k >>= 1) {
        if (threadIdx.x < k) red[threadIdx.x] += red[threadIdx.x + k];
        __syncthreads();
    }
    if (threadIdx.x == 0) dis[i] = rsqrtf(red[0] + 1.0f);
}

__global__ void rowscale_k(const float* __restrict__ xw, const float* __restrict__ dis,
                           float* z, int n, int c) {
    long idx = (long)blockIdx.x * blockDim.x + threadIdx.x;
    if (idx < (long)n * c) z[idx] = dis[idx / c] * xw[idx];
}

// out = act( dis[i]*(t+z) + b[c] );  act: 0=none 1=relu 2=sigmoid
__global__ void gcn_epilogue_k(const float* __restrict__ t, const float* __restrict__ z,
                               const float* __restrict__ dis, const float* __restrict__ b,
                               float* out, int n, int c, int act) {
    long idx = (long)blockIdx.x * blockDim.x + threadIdx.x;
    if (idx < (long)n * c) {
        int i = (int)(idx / c), cc = (int)(idx % c);
        float v = dis[i] * (t[idx] + z[idx]) + b[cc];
        if (act == 1) v = fmaxf(v, 0.0f);
        else if (act == 2) v = 1.0f / (1.0f + expf(-v));
        out[idx] = v;
    }
}

// augment epilogue: C = (A@A) + 2A off-diagonal, 0 on diagonal
__global__ void aug_epilogue_k(float* C, const float* __restrict__ A, int n) {
    long idx = (long)blockIdx.x * blockDim.x + threadIdx.x;
    if (idx < (long)n * n) {
        int i = (int)(idx / n), j = (int)(idx % n);
        C[idx] = (i == j) ? 0.0f : C[idx] + 2.0f * A[idx];
    }
}

// sc[i] = tanh( h[i]·p / ||p|| )
__global__ void score_k(const float* __restrict__ h, const float* __restrict__ p,
                        float* sc, int c) {
    __shared__ float rd[256], rp[256];
    int i = blockIdx.x;
    float d = 0.0f, pn = 0.0f;
    for (int t = threadIdx.x; t < c; t += 256) {
        float pv = p[t];
        d += h[(long)i * c + t] * pv;
        pn += pv * pv;
    }
    rd[threadIdx.x] = d; rp[threadIdx.x] = pn;
    __syncthreads();
    for (int k = 128; k > 0; k >>= 1) {
        if (threadIdx.x < k) { rd[threadIdx.x] += rd[threadIdx.x + k];
                               rp[threadIdx.x] += rp[threadIdx.x + k]; }
        __syncthreads();
    }
    if (threadIdx.x == 0) sc[i] = tanhf(rd[0] / sqrtf(rp[0]));
}

// perm[rank]=i for the k largest scores, ties broken by smaller index (lax.top_k)
__global__ void topk_k(const float* __restrict__ sc, int n, int k, int* perm) {
    int i = blockIdx.x * blockDim.x + threadIdx.x;
    if (i >= n) return;
    float si = sc[i];
    int rank = 0;
    for (int j = 0; j < n; ++j) {
        float sj = sc[j];
        rank += (sj > si) || (sj == si && j < i);
    }
    if (rank < k) perm[rank] = i;
}

__global__ void gather_feat_k(const float* __restrict__ h, const float* __restrict__ sc,
                              const int* __restrict__ perm, float* out, int k, int c) {
    long idx = (long)blockIdx.x * blockDim.x + threadIdx.x;
    if (idx < (long)k * c) {
        int r = (int)(idx / c), cc = (int)(idx % c);
        int pr = perm[r];
        out[idx] = h[(long)pr * c + cc] * sc[pr];
    }
}

__global__ void gather_adj_k(const float* __restrict__ Asrc, const int* __restrict__ perm,
                             float* out, int k, int n) {
    int q = blockIdx.x * 16 + threadIdx.x;
    int r = blockIdx.y * 16 + threadIdx.y;
    if (r < k && q < k)
        out[(long)r * k + q] = Asrc[(long)perm[r] * n + perm[q]];
}

__global__ void copy_k(const float* __restrict__ src, float* dst, long n) {
    long i = (long)blockIdx.x * blockDim.x + threadIdx.x;
    if (i < n) dst[i] = src[i];
}

// out[perm[r]][c] += hin[r][c]   (perm rows unique -> race free)
__global__ void scatter_add_k(float* out, const int* __restrict__ perm,
                              const float* __restrict__ hin, int k, int c) {
    long idx = (long)blockIdx.x * blockDim.x + threadIdx.x;
    if (idx < (long)k * c) {
        int r = (int)(idx / c), cc = (int)(idx % c);
        out[(long)perm[r] * c + cc] += hin[idx];
    }
}

// ---------------------------------------------------------------- host side
extern "C" void kernel_launch(void* const* d_in, const int* in_sizes, int n_in,
                              void* d_out, int out_size, void* d_ws, size_t ws_size,
                              hipStream_t stream) {
    const int N0 = 4096, K1 = 2000, K2 = 1000, K3 = 500;
    const int CIN = 128, CH = 512, COUT = 64, NE = 65536;

    const float* x   = (const float*)d_in[0];
    const int*   ei  = (const int*)  d_in[1];
    const float* Wd0 = (const float*)d_in[2];  const float* bd0 = (const float*)d_in[3];
    const float* Wd1 = (const float*)d_in[4];  const float* bd1 = (const float*)d_in[5];
    const float* Wd2 = (const float*)d_in[6];  const float* bd2 = (const float*)d_in[7];
    const float* Wd3 = (const float*)d_in[8];  const float* bd3 = (const float*)d_in[9];
    const float* p1  = (const float*)d_in[10];
    const float* p2  = (const float*)d_in[11];
    const float* p3  = (const float*)d_in[12];
    const float* Wu0 = (const float*)d_in[13]; const float* bu0 = (const float*)d_in[14];
    const float* Wu1 = (const float*)d_in[15]; const float* bu1 = (const float*)d_in[16];
    const float* Wu2 = (const float*)d_in[17]; const float* bu2 = (const float*)d_in[18];

    float* ws = (float*)d_ws;
    size_t off = 0;
    auto alloc = [&](size_t n) { float* p = ws + off; off += n; return p; };

    float* A0   = alloc((size_t)N0 * N0);      // As[0]
    float* Atmp = alloc((size_t)N0 * N0);      // augment scratch
    float* A1   = alloc((size_t)K1 * K1);      // As[1]
    float* A2   = alloc((size_t)K2 * K2);      // As[2]
    float* A3   = alloc((size_t)K3 * K3);
    float* h0   = alloc((size_t)N0 * CH);      // xs[0]
    float* h1   = alloc((size_t)K1 * CH);      // xs[1]
    float* h2   = alloc((size_t)K2 * CH);      // xs[2]
    float* hp   = alloc((size_t)N0 * CH);      // pooled features
    float* hu   = alloc((size_t)N0 * CH);      // unpool / residual sum
    float* hc   = alloc((size_t)N0 * CH);      // current features
    float* xw   = alloc((size_t)N0 * CH);      // X@W
    float* zb   = alloc((size_t)N0 * CH);      // dis * xw
    float* tb   = alloc((size_t)N0 * CH);      // A @ z
    float* dis  = alloc(N0);
    float* sc   = alloc(N0);
    int*   perm1 = (int*)alloc(K1);
    int*   perm2 = (int*)alloc(K2);
    int*   perm3 = (int*)alloc(K3);
    (void)ws_size; (void)in_sizes; (void)n_in; (void)out_size;

    auto gemm = [&](const float* A, const float* B, float* C, int M, int Nn, int K) {
        dim3 grid((Nn + BN - 1) / BN, (M + BM - 1) / BM);
        gemm_f16wmma<<<grid, 256, 0, stream>>>(A, B, C, M, Nn, K);
    };
    auto ew = [](long n) { return dim3((unsigned)((n + 255) / 256)); };

    // GCN: out = act( D^-1/2 (A+I) D^-1/2 (X@W) + b )
    auto gcn = [&](const float* A, int n, const float* X, int cin,
                   const float* W, const float* b, int cout, float* out, int act) {
        gemm(X, W, xw, n, cout, cin);
        row_dis_k<<<n, 256, 0, stream>>>(A, n, dis);
        long tot = (long)n * cout;
        rowscale_k<<<ew(tot), 256, 0, stream>>>(xw, dis, zb, n, cout);
        gemm(A, zb, tb, n, cout, n);
        gcn_epilogue_k<<<ew(tot), 256, 0, stream>>>(tb, zb, dis, b, out, n, cout, act);
    };
    auto augment = [&](const float* A, int n) {   // Atmp = (A+I)@(A+I) - diag
        gemm(A, A, Atmp, n, n, n);
        aug_epilogue_k<<<ew((long)n * n), 256, 0, stream>>>(Atmp, A, n);
    };
    auto pool = [&](const float* h, int n, const float* p, int k, int* perm,
                    float* Adst) {                // consumes Atmp as augmented adj
        score_k<<<n, 256, 0, stream>>>(h, p, sc, CH);
        topk_k<<<(n + 255) / 256, 256, 0, stream>>>(sc, n, k, perm);
        gather_feat_k<<<ew((long)k * CH), 256, 0, stream>>>(h, sc, perm, hp, k, CH);
        dim3 ga((k + 15) / 16, (k + 15) / 16);
        gather_adj_k<<<ga, dim3(16, 16), 0, stream>>>(Atmp, perm, Adst, k, n);
    };
    auto unpool = [&](const float* res, int n, const int* perm, const float* h, int k) {
        copy_k<<<ew((long)n * CH), 256, 0, stream>>>(res, hu, (long)n * CH);
        scatter_add_k<<<ew((long)k * CH), 256, 0, stream>>>(hu, perm, h, k, CH);
    };

    // ---- build dense symmetric adjacency
    fill_zero_k<<<ew((long)N0 * N0), 256, 0, stream>>>(A0, (long)N0 * N0);
    scatter_edges_k<<<(NE + 255) / 256, 256, 0, stream>>>(ei, A0, NE, N0);

    // ---- initial GCN + relu
    gcn(A0, N0, x, CIN, Wd0, bd0, CH, h0, 1);

    // ---- down path
    augment(A0, N0);
    pool(h0, N0, p1, K1, perm1, A1);
    gcn(A1, K1, hp, CH, Wd1, bd1, CH, h1, 1);

    augment(A1, K1);
    pool(h1, K1, p2, K2, perm2, A2);
    gcn(A2, K2, hp, CH, Wd2, bd2, CH, h2, 1);

    augment(A2, K2);
    pool(h2, K2, p3, K3, perm3, A3);
    gcn(A3, K3, hp, CH, Wd3, bd3, CH, hc, 1);

    // ---- up path
    unpool(h2, K2, perm3, hc, K3);
    gcn(A2, K2, hu, CH, Wu0, bu0, CH, hc, 1);

    unpool(h1, K1, perm2, hc, K2);
    gcn(A1, K1, hu, CH, Wu1, bu1, CH, hc, 1);

    unpool(h0, N0, perm1, hc, K1);
    gcn(A0, N0, hu, CH, Wu2, bu2, COUT, (float*)d_out, 2);  // sigmoid
}
